// PremiseSelectionModel_49151605735567
// MI455X (gfx1250) — matrix operand order repaired and checked by reading, not compiled, not run on hardware
//
#include <hip/hip_runtime.h>
#include <hip/hip_bf16.h>

// ---------------------------------------------------------------------------
// CDNA5 (gfx1250) implementation of the PremiseSelection reference.
// Edge GEMMs (118 GFLOP/layer) run on v_wmma_f32_16x16x32_bf16 with the
// weight (B) tiles staged in LDS via the gfx1250 async-to-LDS DMA path
// (global_load_async_to_lds_b128 + s_wait_asynccnt), double-buffered so the
// next tile's DMA overlaps the current tile's 12 WMMAs. x stays L2-resident
// (25.6 MB bf16 vs 192 MB L2) so per-edge gathers are L2 hits.
// Linear biases provably cancel under batch-stat BN and are dropped.
// ---------------------------------------------------------------------------

typedef __attribute__((ext_vector_type(16))) __bf16 v16bf;
typedef __attribute__((ext_vector_type(8)))  __bf16 v8bf;
typedef __attribute__((ext_vector_type(8)))  float  v8f;

namespace {
constexpr int N_ = 100000;          // nodes
constexpr int E_ = 400000;          // edges (term walks)
constexpr int D_ = 128;             // channels
constexpr int KD = 384;             // 3*D concat feature width (GEMM K dim)
constexpr int OC = 400;             // 384 T|M|B outputs + 1 score + 15 zero pad
constexpr int NT1 = 25;             // N-tiles in stats pass (incl. score tile)
constexpr int NT2 = 24;             // N-tiles in scatter pass (skip score)
constexpr int KKE = 12;             // 384/32 k-steps
constexpr int TILES_E = E_ / 16;    // 25000 (divisible by 8 waves/block)
constexpr int TILES_N = N_ / 16;    // 6250
constexpr int KKN = 4;              // 128/32
constexpr int NTN = 8;              // 128/16
constexpr int WPB = 8;              // waves per 256-thread block (wave32)
constexpr int BROW = 392;           // padded LDS row stride (bf16) edge B tile
constexpr int NROW = 136;           // padded LDS row stride (bf16) node W
constexpr float EPS_ = 1e-5f;
}

__device__ __forceinline__ __bf16 f2bf(float f) {
  unsigned u = __float_as_uint(f);
  unsigned r = u + 0x7fffu + ((u >> 16) & 1u);   // round-to-nearest-even
  unsigned short hs = (unsigned short)(r >> 16);
  __bf16 b;
  __builtin_memcpy(&b, &hs, sizeof(b));
  return b;
}

__device__ __forceinline__ v8f wmma_bf16(v16bf a, v16bf b, v8f c) {
  // (neg_a, A, neg_b, B, c_mod, C, reuse_a, reuse_b)
  return __builtin_amdgcn_wmma_f32_16x16x32_bf16(false, a, false, b,
                                                 (short)0, c, false, false);
}

// ---------------------- async global -> LDS staging ------------------------
// One 16-byte chunk per call; tracked by ASYNCcnt (no VGPR round trip).
__device__ __forceinline__ void async_cp16(const __bf16* gsrc, __bf16* ldst) {
  unsigned lo = (unsigned)(size_t)ldst;   // low 32 bits of LDS-aperture addr
  asm volatile("global_load_async_to_lds_b128 %0, %1, off"
               :: "v"(lo), "v"(gsrc) : "memory");
}
__device__ __forceinline__ void async_wait() {
  asm volatile("s_wait_asynccnt 0x0" ::: "memory");
}

// Stage one 16x384 bf16 B tile (weights for N-tile `nt`) into padded LDS.
// 768 chunks of 16B; 3 per thread (256 threads).
__device__ __forceinline__ void stage_edge_B(const __bf16* __restrict__ WB,
                                             int nt, __bf16* dst, int tid) {
#pragma unroll
  for (int j = 0; j < 3; ++j) {
    int c = tid + 256 * j;            // 0..767
    int row = c / 48;                 // 48 chunks per 384-bf16 row
    int off = (c - row * 48) * 8;     // bf16 units
    async_cp16(WB + (size_t)(nt * 16 + row) * KD + off, dst + row * BROW + off);
  }
}

// Stage the whole 128x128 bf16 node weight matrix into padded LDS.
__device__ __forceinline__ void stage_node_W(const __bf16* __restrict__ W,
                                             __bf16* dst, int tid) {
#pragma unroll
  for (int j = 0; j < 8; ++j) {       // 2048 chunks / 256 threads
    int c = tid + 256 * j;
    int row = c >> 4;                 // 16 chunks per 128-bf16 row
    int off = (c & 15) * 8;
    async_cp16(W + (size_t)row * D_ + off, dst + row * NROW + off);
  }
}

// ---------------------------- elementwise utils ----------------------------

__global__ void k_zero_f32(float* p, int n) {
  int i = blockIdx.x * blockDim.x + threadIdx.x;
  if (i < n) p[i] = 0.f;
}
__global__ void k_zero_i32(int* p, int n) {
  int i = blockIdx.x * blockDim.x + threadIdx.x;
  if (i < n) p[i] = 0;
}
__global__ void k_copy_f32(float* dst, const float* src, int n) {
  int i = blockIdx.x * blockDim.x + threadIdx.x;
  if (i < n) dst[i] = src[i];
}
__global__ void k_cvt_bf16(__bf16* dst, const float* src, int n) {
  int i = blockIdx.x * blockDim.x + threadIdx.x;
  if (i < n) dst[i] = f2bf(src[i]);
}

// Pack per-layer weights into one [OC x KD] bf16 matrix.
// Rows 0..127 = W_T, 128..255 = W_M, 256..383 = W_B, 384 = W_1, 385..399 = 0.
__global__ void k_prep_wb(__bf16* WB, const float* WT, const float* WM,
                          const float* WBm, const float* W1) {
  int i = blockIdx.x * blockDim.x + threadIdx.x;
  if (i >= OC * KD) return;
  int r = i / KD, c = i - r * KD;
  float v;
  if (r < 128)      v = WT[r * KD + c];
  else if (r < 256) v = WM[(r - 128) * KD + c];
  else if (r < 384) v = WBm[(r - 256) * KD + c];
  else if (r == 384) v = W1[c];
  else              v = 0.f;
  WB[i] = f2bf(v);
}

__global__ void k_prep_wtw(__bf16* W, const float* Wtw) {
  int i = blockIdx.x * blockDim.x + threadIdx.x;
  if (i < D_ * D_) W[i] = f2bf(Wtw[i]);
}

// Degree of each node per index array (twi is layer-invariant).
__global__ void k_degrees(const int* twi, int* deg) {
  int i = blockIdx.x * blockDim.x + threadIdx.x;
  if (i >= 3 * E_) return;
  int g = i / E_;
  __hip_atomic_fetch_add(deg + g * N_ + twi[i], 1,
                         __ATOMIC_RELAXED, __HIP_MEMORY_SCOPE_AGENT);
}
__global__ void k_invdeg(const int* deg, float* ivdg, int n) {
  int i = blockIdx.x * blockDim.x + threadIdx.x;
  if (i >= n) return;
  int d = deg[i];
  ivdg[i] = 1.f / (float)(d > 0 ? d : 1);
}

// ------------------------- WMMA fragment loaders ---------------------------
// A (16x32 bf16): lane half h: halfwords 0..7 hold K = kb+8h .. kb+8h+7,
//                 halfwords 8..15 hold K = kb+8h+16 .. kb+8h+23 (ISA 7.12.2).
__device__ __forceinline__ void load_A_edge(const __bf16* __restrict__ xb,
                                            const int* __restrict__ twi,
                                            int tile, int half, int mrow,
                                            v16bf (&A)[KKE]) {
  const int e = tile * 16 + mrow;
  const __bf16* rp[3] = { xb + (size_t)twi[e] * D_,
                          xb + (size_t)twi[E_ + e] * D_,
                          xb + (size_t)twi[2 * E_ + e] * D_ };
#pragma unroll
  for (int kk = 0; kk < KKE; ++kk) {
    const int kb = kk * 32;
    union { v16bf v; v8bf h[2]; } a;
    a.h[0] = *(const v8bf*)(rp[kb >> 7] + ((kb & 127) + 8 * half));
    a.h[1] = *(const v8bf*)(rp[(kb + 16) >> 7] + (((kb + 16) & 127) + 8 * half));
    A[kk] = a.v;
  }
}

__device__ __forceinline__ void load_A_node(const __bf16* __restrict__ mb,
                                            int tile, int half, int mrow,
                                            v16bf (&A)[KKN]) {
  const __bf16* row = mb + (size_t)(tile * 16 + mrow) * D_;
#pragma unroll
  for (int kk = 0; kk < KKN; ++kk) {
    union { v16bf v; v8bf h[2]; } a;
    a.h[0] = *(const v8bf*)(row + kk * 32 + 8 * half);
    a.h[1] = *(const v8bf*)(row + kk * 32 + 16 + 8 * half);
    A[kk] = a.v;
  }
}

// B (32x16 bf16) from padded LDS: N = lane&15, K = kb + 8*half + t (t<8), kb + 16 + 8*half + (t-8) (t>=8).
__device__ __forceinline__ v16bf load_Bs(const __bf16* Bs, int stride,
                                         int kk, int half, int ncol) {
  const __bf16* p = Bs + ncol * stride + kk * 32 + 8 * half;
  union { v16bf v; v8bf h[2]; } b;
  b.h[0] = *(const v8bf*)(p);
  b.h[1] = *(const v8bf*)(p + 16);
  return b.v;
}

// ------------------------- edge GEMM pass 1: stats -------------------------
__global__ void __launch_bounds__(256)
k_edge_pass1(const __bf16* __restrict__ xb, const __bf16* __restrict__ WB,
             const int* __restrict__ twi, float* __restrict__ score,
             float* __restrict__ ssum, float* __restrict__ ssq) {
  __shared__ __bf16 Bs[2][16 * BROW];
  __shared__ float lsum[OC];
  __shared__ float lsq[OC];
  const int tid = threadIdx.x;
  for (int i = tid; i < OC; i += 256) { lsum[i] = 0.f; lsq[i] = 0.f; }
  __syncthreads();

  const int lane = tid & 31, wave = tid >> 5;
  const int half = lane >> 4, mrow = lane & 15;

  for (int tb = blockIdx.x * WPB; tb < TILES_E; tb += gridDim.x * WPB) {
    const int tile = tb + wave;              // TILES_E % WPB == 0 -> in range
    stage_edge_B(WB, 0, Bs[0], tid);         // overlap DMA with A gather
    v16bf A[KKE];
    load_A_edge(xb, twi, tile, half, mrow, A);
    async_wait();
    __syncthreads();
    int buf = 0;
#pragma unroll 1
    for (int nt = 0; nt < NT1; ++nt) {
      if (nt + 1 < NT1) stage_edge_B(WB, nt + 1, Bs[buf ^ 1], tid);
      v8f acc = {0.f, 0.f, 0.f, 0.f, 0.f, 0.f, 0.f, 0.f};
#pragma unroll
      for (int kk = 0; kk < KKE; ++kk)
        acc = wmma_bf16(A[kk], load_Bs(Bs[buf], BROW, kk, half, mrow), acc);
      float s = 0.f, q = 0.f;
#pragma unroll
      for (int r = 0; r < 8; ++r) { float v = acc[r]; s += v; q += v * v; }
      const int c = nt * 16 + mrow;
      __hip_atomic_fetch_add(&lsum[c], s, __ATOMIC_RELAXED,
                             __HIP_MEMORY_SCOPE_WORKGROUP);
      __hip_atomic_fetch_add(&lsq[c], q, __ATOMIC_RELAXED,
                             __HIP_MEMORY_SCOPE_WORKGROUP);
      if (nt == NT1 - 1 && mrow == 0) {      // channel 384 = raw score
#pragma unroll
        for (int r = 0; r < 8; ++r) score[tile * 16 + 8 * half + r] = acc[r];
      }
      async_wait();                          // staged tile landed in LDS
      __syncthreads();                       // all waves done with Bs[buf]
      buf ^= 1;
    }
  }
  for (int i = tid; i < OC; i += 256) {
    __hip_atomic_fetch_add(&ssum[i], lsum[i], __ATOMIC_RELAXED,
                           __HIP_MEMORY_SCOPE_AGENT);
    __hip_atomic_fetch_add(&ssq[i], lsq[i], __ATOMIC_RELAXED,
                           __HIP_MEMORY_SCOPE_AGENT);
  }
}

// Fold BN into scale/shift per channel (bias cancels under batch-stat BN).
__global__ void k_fin_edge(const float* ssum, const float* ssq,
                           const float* gT, const float* beT,
                           const float* gM, const float* beM,
                           const float* gB, const float* beB,
                           const float* g1, const float* be1,
                           float* scale, float* shift) {
  int c = blockIdx.x * blockDim.x + threadIdx.x;
  if (c >= OC) return;
  if (c > 384) { scale[c] = 0.f; shift[c] = 0.f; return; }
  const float inv = 1.f / (float)E_;
  float mu = ssum[c] * inv;
  float var = ssq[c] * inv - mu * mu;
  var = var > 0.f ? var : 0.f;
  float rs = rsqrtf(var + EPS_);
  float g, be;
  if (c < 128)      { g = gT[c];       be = beT[c]; }
  else if (c < 256) { g = gM[c - 128]; be = beM[c - 128]; }
  else if (c < 384) { g = gB[c - 256]; be = beB[c - 256]; }
  else              { g = g1[0];       be = be1[0]; }
  scale[c] = g * rs;
  shift[c] = be - mu * g * rs;
}

__global__ void k_fin_node(const float* ssum, const float* ssq,
                           const float* g, const float* be,
                           float* scale, float* shift) {
  int c = blockIdx.x * blockDim.x + threadIdx.x;
  if (c >= D_) return;
  const float inv = 1.f / (float)N_;
  float mu = ssum[c] * inv;
  float var = ssq[c] * inv - mu * mu;
  var = var > 0.f ? var : 0.f;
  float rs = rsqrtf(var + EPS_);
  scale[c] = g[c] * rs;
  shift[c] = be[c] - mu * g[c] * rs;
}

// ------------------------- softmax over E (3 small passes) -----------------
__global__ void k_att_max(const float* score, const float* bsc,
                          const float* bsh, float* gms) {
  __shared__ float red[256];
  const float sc = bsc[384], sh = bsh[384];
  float m = 0.f;   // relu output >= 0
  for (int i = blockIdx.x * blockDim.x + threadIdx.x; i < E_;
       i += gridDim.x * blockDim.x) {
    float s = score[i] * sc + sh;
    s = s > 0.f ? s : 0.f;
    m = s > m ? s : m;
  }
  red[threadIdx.x] = m; __syncthreads();
  for (int s = 128; s > 0; s >>= 1) {
    if (threadIdx.x < s) red[threadIdx.x] = fmaxf(red[threadIdx.x], red[threadIdx.x + s]);
    __syncthreads();
  }
  if (threadIdx.x == 0)  // non-negative floats compare correctly as ints
    __hip_atomic_fetch_max((int*)gms, __float_as_int(red[0]),
                           __ATOMIC_RELAXED, __HIP_MEMORY_SCOPE_AGENT);
}

__global__ void k_att_sum(const float* score, const float* bsc,
                          const float* bsh, float* gms) {
  __shared__ float red[256];
  const float sc = bsc[384], sh = bsh[384];
  const float gmax = gms[0];
  float acc = 0.f;
  for (int i = blockIdx.x * blockDim.x + threadIdx.x; i < E_;
       i += gridDim.x * blockDim.x) {
    float s = score[i] * sc + sh;
    s = s > 0.f ? s : 0.f;
    acc += __expf(s - gmax);
  }
  red[threadIdx.x] = acc; __syncthreads();
  for (int s = 128; s > 0; s >>= 1) {
    if (threadIdx.x < s) red[threadIdx.x] += red[threadIdx.x + s];
    __syncthreads();
  }
  if (threadIdx.x == 0)
    __hip_atomic_fetch_add(gms + 1, red[0], __ATOMIC_RELAXED,
                           __HIP_MEMORY_SCOPE_AGENT);
}

// Per-edge scatter weight: w_g[e] = att[e] / deg_g[idx_g[e]]  (g = 0,1,2)
__global__ void k_att_w(const float* score, const float* bsc, const float* bsh,
                        const float* gms, const int* twi, const float* ivdg,
                        float* wbuf) {
  int i = blockIdx.x * blockDim.x + threadIdx.x;
  if (i >= E_) return;
  const float sc = bsc[384], sh = bsh[384];
  float s = score[i] * sc + sh;
  s = s > 0.f ? s : 0.f;
  float a = __expf(s - gms[0]) / gms[1];
#pragma unroll
  for (int g = 0; g < 3; ++g)
    wbuf[g * E_ + i] = a * ivdg[g * N_ + twi[g * E_ + i]];
}

// ---------------- edge GEMM pass 2: BN+ReLU+att, scatter-add ---------------
__global__ void __launch_bounds__(256)
k_edge_pass2(const __bf16* __restrict__ xb, const __bf16* __restrict__ WB,
             const int* __restrict__ twi, const float* __restrict__ wbuf,
             const float* __restrict__ bsc, const float* __restrict__ bsh,
             float* __restrict__ macc) {
  __shared__ __bf16 Bs[2][16 * BROW];
  const int tid = threadIdx.x, lane = tid & 31, wave = tid >> 5;
  const int half = lane >> 4, mrow = lane & 15;

  for (int tb = blockIdx.x * WPB; tb < TILES_E; tb += gridDim.x * WPB) {
    const int tile = tb + wave;
    stage_edge_B(WB, 0, Bs[0], tid);
    v16bf A[KKE];
    load_A_edge(xb, twi, tile, half, mrow, A);
    const int ebase = tile * 16 + 8 * half;  // D layout: M = r + 8*half
    async_wait();
    __syncthreads();
    int buf = 0;
#pragma unroll 1
    for (int nt = 0; nt < NT2; ++nt) {
      if (nt + 1 < NT2) stage_edge_B(WB, nt + 1, Bs[buf ^ 1], tid);
      v8f acc = {0.f, 0.f, 0.f, 0.f, 0.f, 0.f, 0.f, 0.f};
#pragma unroll
      for (int kk = 0; kk < KKE; ++kk)
        acc = wmma_bf16(A[kk], load_Bs(Bs[buf], BROW, kk, half, mrow), acc);
      const int g = nt >> 3;                 // 0:T/i0  1:M/i1  2:B/i2
      const int c = nt * 16 + mrow;
      const float sc = bsc[c], sh = bsh[c];
      const int col = c & (D_ - 1);
#pragma unroll
      for (int r = 0; r < 8; ++r) {
        const int e2 = g * E_ + ebase + r;
        float v = acc[r] * sc + sh;
        v = v > 0.f ? v : 0.f;
        __hip_atomic_fetch_add(macc + (size_t)twi[e2] * D_ + col,
                               v * wbuf[e2],
                               __ATOMIC_RELAXED, __HIP_MEMORY_SCOPE_AGENT);
      }
      async_wait();
      __syncthreads();
      buf ^= 1;
    }
  }
}

// -------------------------- node GEMM (W_tw MLP) ---------------------------
__global__ void __launch_bounds__(256)
k_node_pass1(const __bf16* __restrict__ mb, const __bf16* __restrict__ W,
             float* __restrict__ ssum, float* __restrict__ ssq) {
  __shared__ __bf16 Ws[D_ * NROW];
  __shared__ float lsum[D_];
  __shared__ float lsq[D_];
  const int tid = threadIdx.x;
  stage_node_W(W, Ws, tid);
  if (tid < D_) { lsum[tid] = 0.f; lsq[tid] = 0.f; }
  async_wait();
  __syncthreads();

  const int lane = tid & 31, wave = tid >> 5;
  const int half = lane >> 4, mrow = lane & 15;
  for (int tile = blockIdx.x * WPB + wave; tile < TILES_N;
       tile += gridDim.x * WPB) {
    v16bf A[KKN];
    load_A_node(mb, tile, half, mrow, A);
#pragma unroll 1
    for (int nt = 0; nt < NTN; ++nt) {
      v8f acc = {0.f, 0.f, 0.f, 0.f, 0.f, 0.f, 0.f, 0.f};
#pragma unroll
      for (int kk = 0; kk < KKN; ++kk)
        acc = wmma_bf16(A[kk], load_Bs(Ws + nt * 16 * NROW, NROW,
                                       kk, half, mrow), acc);
      float s = 0.f, q = 0.f;
#pragma unroll
      for (int r = 0; r < 8; ++r) { float v = acc[r]; s += v; q += v * v; }
      const int c = nt * 16 + mrow;
      __hip_atomic_fetch_add(&lsum[c], s, __ATOMIC_RELAXED,
                             __HIP_MEMORY_SCOPE_WORKGROUP);
      __hip_atomic_fetch_add(&lsq[c], q, __ATOMIC_RELAXED,
                             __HIP_MEMORY_SCOPE_WORKGROUP);
    }
  }
  __syncthreads();
  if (tid < D_) {
    __hip_atomic_fetch_add(&ssum[tid], lsum[tid], __ATOMIC_RELAXED,
                           __HIP_MEMORY_SCOPE_AGENT);
    __hip_atomic_fetch_add(&ssq[tid], lsq[tid], __ATOMIC_RELAXED,
                           __HIP_MEMORY_SCOPE_AGENT);
  }
}

__global__ void __launch_bounds__(256)
k_node_pass2(const __bf16* __restrict__ mb, const __bf16* __restrict__ W,
             const float* __restrict__ nsc, const float* __restrict__ nsh,
             float* __restrict__ out) {
  __shared__ __bf16 Ws[D_ * NROW];
  const int tid = threadIdx.x;
  stage_node_W(W, Ws, tid);
  async_wait();
  __syncthreads();

  const int lane = tid & 31, wave = tid >> 5;
  const int half = lane >> 4, mrow = lane & 15;
  for (int tile = blockIdx.x * WPB + wave; tile < TILES_N;
       tile += gridDim.x * WPB) {
    v16bf A[KKN];
    load_A_node(mb, tile, half, mrow, A);
#pragma unroll 1
    for (int nt = 0; nt < NTN; ++nt) {
      v8f acc = {0.f, 0.f, 0.f, 0.f, 0.f, 0.f, 0.f, 0.f};
#pragma unroll
      for (int kk = 0; kk < KKN; ++kk)
        acc = wmma_bf16(A[kk], load_Bs(Ws + nt * 16 * NROW, NROW,
                                       kk, half, mrow), acc);
      const int c = nt * 16 + mrow;
      const float sc = nsc[c], sh = nsh[c];
#pragma unroll
      for (int r = 0; r < 8; ++r) {   // residual: x += relu(BN(h))
        float v = acc[r] * sc + sh;
        v = v > 0.f ? v : 0.f;
        const size_t o = (size_t)(tile * 16 + 8 * half + r) * D_ + c;
        out[o] = out[o] + v;          // each (n,c) written exactly once
      }
    }
  }
}

// -------------------------------- launcher ---------------------------------
extern "C" void kernel_launch(void* const* d_in, const int* in_sizes, int n_in,
                              void* d_out, int out_size, void* d_ws, size_t ws_size,
                              hipStream_t stream) {
  (void)in_sizes; (void)n_in; (void)out_size; (void)ws_size;
  const float* x    = (const float*)d_in[0];
  const int*   twi  = (const int*)  d_in[1];
  const float* W_T  = (const float*)d_in[2];
  const float* W_M  = (const float*)d_in[3];
  const float* W_Bm = (const float*)d_in[4];
  const float* W_1  = (const float*)d_in[5];
  const float* W_tw = (const float*)d_in[6];
  // biases (indices 7,10,13,16,19) cancel under batch-stat BN -> unused
  const float* g_T  = (const float*)d_in[8];
  const float* be_T = (const float*)d_in[9];
  const float* g_M  = (const float*)d_in[11];
  const float* be_M = (const float*)d_in[12];
  const float* g_B  = (const float*)d_in[14];
  const float* be_B = (const float*)d_in[15];
  const float* g_1  = (const float*)d_in[17];
  const float* be_1 = (const float*)d_in[18];
  const float* g_tw = (const float*)d_in[20];
  const float* be_tw= (const float*)d_in[21];
  float* out = (float*)d_out;

  char* base = (char*)d_ws;
  size_t off = 0;
  auto take = [&](size_t bytes) -> char* {
    char* p = base + off;
    off += (bytes + 255) & ~(size_t)255;
    return p;
  };
  __bf16* xb   = (__bf16*)take((size_t)N_ * D_ * 2);   // bf16 node features
  __bf16* mb   = (__bf16*)take((size_t)N_ * D_ * 2);   // bf16 aggregated msg
  __bf16* WB   = (__bf16*)take((size_t)OC * KD * 2);   // packed edge weights
  __bf16* WTW  = (__bf16*)take((size_t)D_ * D_ * 2);   // node weights
  float*  macc = (float*) take((size_t)N_ * D_ * 4);   // scatter accumulator
  float*  scr  = (float*) take((size_t)E_ * 4);        // raw scores
  float*  wbuf = (float*) take((size_t)3 * E_ * 4);    // att/deg per edge
  int*    deg  = (int*)   take((size_t)3 * N_ * 4);
  float*  ivdg = (float*) take((size_t)3 * N_ * 4);
  float*  ssum = (float*) take(OC * 4);
  float*  ssq  = (float*) take(OC * 4);
  float*  bsc  = (float*) take(OC * 4);
  float*  bsh  = (float*) take(OC * 4);
  float*  nsum = (float*) take(D_ * 4);
  float*  nsq  = (float*) take(D_ * 4);
  float*  nsc  = (float*) take(D_ * 4);
  float*  nsh  = (float*) take(D_ * 4);
  float*  gms  = (float*) take(2 * 4);                 // [max, sumexp]

  auto gr = [](int n) { return dim3((unsigned)((n + 255) / 256)); };
  const dim3 blk(256);
  const dim3 gE(1024), gN(512);

  // one-time: residual seed + degrees
  k_copy_f32<<<gr(N_ * D_), blk, 0, stream>>>(out, x, N_ * D_);
  k_zero_i32<<<gr(3 * N_), blk, 0, stream>>>(deg, 3 * N_);
  k_degrees <<<gr(3 * E_), blk, 0, stream>>>(twi, deg);
  k_invdeg  <<<gr(3 * N_), blk, 0, stream>>>(deg, ivdg, 3 * N_);

  for (int k = 0; k < 2; ++k) {
    k_cvt_bf16<<<gr(N_ * D_), blk, 0, stream>>>(xb, out, N_ * D_);
    k_prep_wb <<<gr(OC * KD), blk, 0, stream>>>(WB,
        W_T + (size_t)k * D_ * KD, W_M + (size_t)k * D_ * KD,
        W_Bm + (size_t)k * D_ * KD, W_1 + (size_t)k * KD);
    k_prep_wtw<<<gr(D_ * D_), blk, 0, stream>>>(WTW, W_tw + (size_t)k * D_ * D_);
    k_zero_f32<<<gr(OC), blk, 0, stream>>>(ssum, OC);
    k_zero_f32<<<gr(OC), blk, 0, stream>>>(ssq, OC);
    k_zero_f32<<<1, blk, 0, stream>>>(gms, 2);
    k_zero_f32<<<gr(N_ * D_), blk, 0, stream>>>(macc, N_ * D_);

    k_edge_pass1<<<gE, blk, 0, stream>>>(xb, WB, twi, scr, ssum, ssq);
    k_fin_edge<<<2, blk, 0, stream>>>(ssum, ssq,
        g_T + k * D_, be_T + k * D_, g_M + k * D_, be_M + k * D_,
        g_B + k * D_, be_B + k * D_, g_1 + k, be_1 + k, bsc, bsh);
    k_att_max<<<512, blk, 0, stream>>>(scr, bsc, bsh, gms);
    k_att_sum<<<512, blk, 0, stream>>>(scr, bsc, bsh, gms);
    k_att_w  <<<gr(E_), blk, 0, stream>>>(scr, bsc, bsh, gms, twi, ivdg, wbuf);
    k_edge_pass2<<<gE, blk, 0, stream>>>(xb, WB, twi, wbuf, bsc, bsh, macc);

    k_cvt_bf16<<<gr(N_ * D_), blk, 0, stream>>>(mb, macc, N_ * D_);
    k_zero_f32<<<1, blk, 0, stream>>>(nsum, D_);
    k_zero_f32<<<1, blk, 0, stream>>>(nsq, D_);
    k_node_pass1<<<gN, blk, 0, stream>>>(mb, WTW, nsum, nsq);
    k_fin_node<<<1, blk, 0, stream>>>(nsum, nsq, g_tw + k * D_, be_tw + k * D_,
                                      nsc, nsh);
    k_node_pass2<<<gN, blk, 0, stream>>>(mb, WTW, nsc, nsh, out);
  }
}
